// PointnetSAModuleMSG_67508295958601
// MI455X (gfx1250) — matrix operand order repaired and checked by reading, not compiled, not run on hardware
//
#include <hip/hip_runtime.h>
#include <hip/hip_fp16.h>
#include <math.h>

// ---------------------------------------------------------------- types
typedef __attribute__((ext_vector_type(16))) _Float16 v16h;
typedef __attribute__((ext_vector_type(8)))  _Float16 v8h;
typedef __attribute__((ext_vector_type(8)))  float    v8f;

union V16 { v16h v; v8h h[2]; };

// ---------------------------------------------------------------- problem constants
#define PB   4          // batch
#define PN   8192       // points
#define PS   1024       // npoint (FPS samples)
#define PCIN 32         // input feature channels
#define DOUT 128
#define BNEPS 1e-5f

// ================================================================ FPS
// One block per batch. 1024 threads, 8 points each, cached in registers.
// Writes new_xyz (gathered coords) directly to out_xyz.
__global__ __launch_bounds__(1024)
void fps_kernel(const float* __restrict__ xyz, float* __restrict__ out_xyz)
{
    const int b = blockIdx.x;
    const int t = threadIdx.x;
    const float* P = xyz + (size_t)b * PN * 3;

    float px[8], py[8], pz[8], d[8];
#pragma unroll
    for (int r = 0; r < 8; ++r) {
        int i = t * 8 + r;
        px[r] = P[i * 3 + 0];
        py[r] = P[i * 3 + 1];
        pz[r] = P[i * 3 + 2];
        d[r]  = 1e10f;
    }

    __shared__ float swv[32];
    __shared__ int   swi[32];
    __shared__ int   s_last;

    int last = 0;
    for (int it = 0; it < PS; ++it) {
        if (t == 0) {
            float* o = out_xyz + ((size_t)b * PS + it) * 3;
            o[0] = P[last * 3 + 0];
            o[1] = P[last * 3 + 1];
            o[2] = P[last * 3 + 2];
        }
        const float lx = P[last * 3 + 0];
        const float ly = P[last * 3 + 1];
        const float lz = P[last * 3 + 2];

        float bv = -1.0f; int bi = 0;
#pragma unroll
        for (int r = 0; r < 8; ++r) {
            float dx = px[r] - lx, dy = py[r] - ly, dz = pz[r] - lz;
            float dd = dx * dx + dy * dy + dz * dz;
            d[r] = fminf(d[r], dd);
            int i = t * 8 + r;
            if (d[r] > bv || (d[r] == bv && i < bi)) { bv = d[r]; bi = i; }
        }
        // wave32 reduce (argmax, prefer lower index on tie)
#pragma unroll
        for (int off = 16; off > 0; off >>= 1) {
            float ov = __shfl_down(bv, off, 32);
            int   oi = __shfl_down(bi, off, 32);
            if (ov > bv || (ov == bv && oi < bi)) { bv = ov; bi = oi; }
        }
        if ((t & 31) == 0) { swv[t >> 5] = bv; swi[t >> 5] = bi; }
        __syncthreads();
        if (t < 32) {
            bv = swv[t]; bi = swi[t];
#pragma unroll
            for (int off = 16; off > 0; off >>= 1) {
                float ov = __shfl_down(bv, off, 32);
                int   oi = __shfl_down(bi, off, 32);
                if (ov > bv || (ov == bv && oi < bi)) { bv = ov; bi = oi; }
            }
            if (t == 0) s_last = bi;
        }
        __syncthreads();
        last = s_last;
    }
}

// ================================================================ ball query
// One thread per (b,s) query; first-k-by-index within radius, pad with first.
__global__ __launch_bounds__(256)
void ballquery_kernel(const float* __restrict__ xyz, const float* __restrict__ newxyz,
                      int* __restrict__ idx, float r2, int K)
{
    const int t = blockIdx.x * blockDim.x + threadIdx.x;  // b*PS + s
    if (t >= PB * PS) return;
    const int b = t / PS;
    const float* P = xyz + (size_t)b * PN * 3;
    const float qx = newxyz[t * 3 + 0];
    const float qy = newxyz[t * 3 + 1];
    const float qz = newxyz[t * 3 + 2];

    int* row = idx + (size_t)t * K;
    int cnt = 0, first = 0;
    for (int n = 0; n < PN; ++n) {
        float dx = P[n * 3 + 0] - qx;
        float dy = P[n * 3 + 1] - qy;
        float dz = P[n * 3 + 2] - qz;
        float d2 = dx * dx + dy * dy + dz * dz;
        if (d2 < r2) {
            if (cnt == 0) first = n;
            row[cnt++] = n;
            if (cnt == K) break;
        }
    }
    for (int k = cnt; k < K; ++k) row[k] = first;  // cnt==0 -> all zeros
}

// ================================================================ grouping
// Builds transposed f16 operands: XTin (Ntot x 64): [rel(3), feat(32), pad],
// XTxyz (Ntot x 32): [dist, rel(3), ctr(3), nbr(3), pad].
__global__ __launch_bounds__(256)
void group_kernel(const float* __restrict__ xyz, const float* __restrict__ feats,
                  const float* __restrict__ newxyz, const int* __restrict__ idx,
                  _Float16* __restrict__ XTin, _Float16* __restrict__ XTxyz,
                  int K, int Ntot)
{
    const int j = blockIdx.x * blockDim.x + threadIdx.x;  // (b*PS+s)*K + k
    if (j >= Ntot) return;
    const int bs = j / K;
    const int b  = bs / PS;
    const int n  = idx[j];

    const float* P = xyz + ((size_t)b * PN + n) * 3;
    const float nx = P[0], ny = P[1], nz = P[2];
    const float cx = newxyz[bs * 3 + 0];
    const float cy = newxyz[bs * 3 + 1];
    const float cz = newxyz[bs * 3 + 2];
    const float rx = nx - cx, ry = ny - cy, rz = nz - cz;

    _Float16* xi = XTin + (size_t)j * 64;
    xi[0] = (_Float16)rx; xi[1] = (_Float16)ry; xi[2] = (_Float16)rz;
    const float* F = feats + (size_t)b * PCIN * PN + n;
#pragma unroll
    for (int c = 0; c < PCIN; ++c) xi[3 + c] = (_Float16)F[(size_t)c * PN];
#pragma unroll
    for (int c = 3 + PCIN; c < 64; ++c) xi[c] = (_Float16)0.0f;

    const float dist = sqrtf(fmaxf(rx * rx + ry * ry + rz * rz, 1e-12f));
    _Float16* xs = XTxyz + (size_t)j * 32;
    xs[0] = (_Float16)dist;
    xs[1] = (_Float16)rx; xs[2] = (_Float16)ry; xs[3] = (_Float16)rz;
    xs[4] = (_Float16)cx; xs[5] = (_Float16)cy; xs[6] = (_Float16)cz;
    xs[7] = (_Float16)nx; xs[8] = (_Float16)ny; xs[9] = (_Float16)nz;
#pragma unroll
    for (int c = 10; c < 32; ++c) xs[c] = (_Float16)0.0f;
}

// ================================================================ WMMA GEMM (KPAD compile-time)
// Y[M x Ncols] = Wf[M x KPAD] * XT^T  (+ bias), XT stored (Ncols x KPAD) f16.
// Block = 4 waves; each wave computes 4 tiles of 16x16.
//  - A strip (16 x KPAD, shared by all 4 waves) staged in LDS once, read via ds_load_b128.
//  - B fragments double-buffered in registers: loads for step s+1 issued before
//    the WMMAs of step s, so the matrix pipe overlaps VMEM.
//  - Bias preloaded as a v8f vector (branchless epilogue).
template <int KPAD>
__global__ __launch_bounds__(128)
void wmma_gemm(const _Float16* __restrict__ Wf, const _Float16* __restrict__ XT,
               const float* __restrict__ bias, float* __restrict__ Y,
               int M, int Ncols)
{
    constexpr int NK = KPAD / 32;
    __shared__ _Float16 sA[16 * KPAD];

    const int lane = threadIdx.x & 31;
    const int wave = threadIdx.x >> 5;
    const int m0   = blockIdx.y * 16;
    const int n0   = blockIdx.x * 256 + wave * 64;
    const int half = lane >> 4;
    const int l16  = lane & 15;

    // ---- stage the A strip (weights) into LDS cooperatively
    {
        constexpr int CHUNKS = (16 * KPAD) / 8;   // v8h chunks
        const v8h* src = (const v8h*)(Wf + (size_t)m0 * KPAD);
        v8h* dst = (v8h*)sA;
#pragma unroll
        for (int c = threadIdx.x; c < CHUNKS; c += 128) dst[c] = src[c];
    }

    // ---- bias for this lane's 8 output rows: contiguous 32B-aligned run
    v8f bvec = {};
    if (bias) bvec = *(const v8f*)(bias + m0 + 8 * half);

    __syncthreads();

    // ---- per-wave B row pointers (already offset by the K-half for this lane)
    const _Float16* xrow[4];
#pragma unroll
    for (int t = 0; t < 4; ++t) {
        xrow[t] = XT + (size_t)(n0 + t * 16 + l16) * KPAD + 16 * half;
        __builtin_prefetch(xrow[t], 0, 0);        // global_prefetch_b8
    }

    v8f acc[4] = {};
    const _Float16* sArow = sA + l16 * KPAD;

    // ---- double-buffered B fragments (parity index is compile-time after unroll)
    V16 bb[2][4];
#pragma unroll
    for (int t = 0; t < 4; ++t) {
        bb[0][t].h[0] = *(const v8h*)(xrow[t]);
        bb[0][t].h[1] = *(const v8h*)(xrow[t] + 8);
    }

#pragma unroll
    for (int s = 0; s < NK; ++s) {
        const int kb  = s * 32;
        const int cur = s & 1;
        const int nxt = cur ^ 1;

        // A 16x32 fragment from LDS: elems 0..15 -> K = kb + 16*half + e
        V16 a;
        a.h[0] = *(const v8h*)(sArow + kb + 16 * half);
        a.h[1] = *(const v8h*)(sArow + kb + 16 * half + 8);

        if (s + 1 < NK) {
#pragma unroll
            for (int t = 0; t < 4; ++t) {
                bb[nxt][t].h[0] = *(const v8h*)(xrow[t] + kb + 32);
                bb[nxt][t].h[1] = *(const v8h*)(xrow[t] + kb + 40);
            }
        }

#pragma unroll
        for (int t = 0; t < 4; ++t) {
            acc[t] = __builtin_amdgcn_wmma_f32_16x16x32_f16(
                false, a.v, false, bb[cur][t].v, (short)0, acc[t], false, false);
        }
    }

    // ---- epilogue: C/D layout m = m0 + r + 8*half, n = lane&15 (branchless)
#pragma unroll
    for (int t = 0; t < 4; ++t) {
        const int n = n0 + t * 16 + l16;
#pragma unroll
        for (int r = 0; r < 8; ++r) {
            const int m = m0 + r + 8 * half;
            Y[(size_t)m * Ncols + n] = acc[t][r] + bvec[r];
        }
    }
}

// ================================================================ BN stats (per channel row of Y)
__global__ __launch_bounds__(256)
void stats_kernel(const float* __restrict__ Y, float* __restrict__ mean,
                  float* __restrict__ var, int Ncols)
{
    const int m = blockIdx.x;
    const float* row = Y + (size_t)m * Ncols;
    float s = 0.f, ss = 0.f;
    for (int n = threadIdx.x; n < Ncols; n += 256) {
        float v = row[n];
        s += v; ss += v * v;
    }
    __shared__ float s1[256], s2[256];
    s1[threadIdx.x] = s; s2[threadIdx.x] = ss;
    __syncthreads();
    for (int st = 128; st > 0; st >>= 1) {
        if (threadIdx.x < st) {
            s1[threadIdx.x] += s1[threadIdx.x + st];
            s2[threadIdx.x] += s2[threadIdx.x + st];
        }
        __syncthreads();
    }
    if (threadIdx.x == 0) {
        float mu = s1[0] / (float)Ncols;
        mean[m] = mu;
        var[m]  = s2[0] / (float)Ncols - mu * mu;
    }
}

// ================================================================ BN + activation -> transposed f16
__global__ __launch_bounds__(256)
void bnact_kernel(const float* __restrict__ Y, const float* __restrict__ mean,
                  const float* __restrict__ var, const float* __restrict__ g,
                  const float* __restrict__ be, _Float16* __restrict__ XT,
                  int M, int Ncols, int ld, int rowOff,
                  float negSlope, int doBN, int doAct)
{
    const size_t i = (size_t)blockIdx.x * 256 + threadIdx.x;
    const size_t total = (size_t)M * Ncols;
    if (i >= total) return;
    const int m = (int)(i / Ncols);
    const int n = (int)(i % Ncols);
    float v = Y[i];
    if (doBN) v = (v - mean[m]) * rsqrtf(var[m] + BNEPS) * g[m] + be[m];
    if (doAct) v = (v > 0.f) ? v : v * negSlope;
    XT[(size_t)n * ld + rowOff + m] = (_Float16)v;
}

// ================================================================ softmax over K + weighted agg
// logits: (192 x Ntot) f32 ; XTf: (Ntot x 192) f16 ; aggXT: (BS x 192) f16.
__global__ __launch_bounds__(256)
void attagg_kernel(const float* __restrict__ logits, const _Float16* __restrict__ XTf,
                   _Float16* __restrict__ aggXT, int Ncols, int K)
{
    const int t = blockIdx.x * blockDim.x + threadIdx.x;  // bs*192 + o
    if (t >= PB * PS * 192) return;
    const int bs = t / 192;
    const int o  = t % 192;
    const float* L = logits + (size_t)o * Ncols + (size_t)bs * K;

    float mx = -1e30f;
    for (int k = 0; k < K; ++k) mx = fmaxf(mx, L[k]);
    float sum = 0.f;
    for (int k = 0; k < K; ++k) sum += __expf(L[k] - mx);
    const float inv = 1.0f / sum;
    float agg = 0.f;
    for (int k = 0; k < K; ++k) {
        float a = __expf(L[k] - mx) * inv;
        agg += a * (float)XTf[((size_t)bs * K + k) * 192 + o];
    }
    aggXT[(size_t)bs * 192 + o] = (_Float16)agg;
}

// ================================================================ final writeout (B,256,S)
__global__ __launch_bounds__(256)
void writeout_kernel(const float* __restrict__ Yf, float* __restrict__ outF, int coff)
{
    const int t = blockIdx.x * blockDim.x + threadIdx.x;  // o*BS + n
    if (t >= DOUT * PB * PS) return;
    const int o = t / (PB * PS);
    const int n = t % (PB * PS);
    const int b = n / PS;
    const int s = n % PS;
    outF[((size_t)b * 256 + coff + o) * PS + s] = Yf[t];
}

// ================================================================ weight f32 -> padded f16 (M x Kpad)
__global__ __launch_bounds__(256)
void wconv_kernel(const float* __restrict__ W, _Float16* __restrict__ Wf,
                  int M, int K, int Kpad)
{
    const int t = blockIdx.x * blockDim.x + threadIdx.x;
    if (t >= M * Kpad) return;
    const int m = t / Kpad;
    const int k = t % Kpad;
    Wf[t] = (k < K) ? (_Float16)W[m * K + k] : (_Float16)0.0f;
}

// ================================================================ host GEMM dispatch
static void launch_gemm(const _Float16* Wf, const _Float16* XT, const float* bias,
                        float* Y, int M, int Kpad, int Ncols, hipStream_t s)
{
    dim3 g(Ncols / 256, M / 16), b(128);
    switch (Kpad) {
    case 32:  wmma_gemm<32 ><<<g, b, 0, s>>>(Wf, XT, bias, Y, M, Ncols); break;
    case 64:  wmma_gemm<64 ><<<g, b, 0, s>>>(Wf, XT, bias, Y, M, Ncols); break;
    case 96:  wmma_gemm<96 ><<<g, b, 0, s>>>(Wf, XT, bias, Y, M, Ncols); break;
    case 128: wmma_gemm<128><<<g, b, 0, s>>>(Wf, XT, bias, Y, M, Ncols); break;
    case 192: wmma_gemm<192><<<g, b, 0, s>>>(Wf, XT, bias, Y, M, Ncols); break;
    default:  break;
    }
}

// ================================================================ host
extern "C" void kernel_launch(void* const* d_in, const int* in_sizes, int n_in,
                              void* d_out, int out_size, void* d_ws, size_t ws_size,
                              hipStream_t stream)
{
    if (n_in < 27) return;
    const float* xyz   = (const float*)d_in[0];
    const float* feats = (const float*)d_in[1];
    // mlp0 L0: 2..5 | mlp0 L1: 6..9 | mlp1 L0: 10..13 | mlp1 L1: 14..17
    // att: Wxyz 18, bxyz 19, Wfc 20, Wm 21, bm 22, g 23, be 24 | fuse: W 25, b 26

    const int    KS[2]   = {16, 32};
    const float  RAD[2]  = {0.1f, 0.2f};
    const int    CMID[2] = {64, 96};
    const int    L0W[2]  = {2, 10};   // first-layer param base index per scale
    const int    L1W[2]  = {6, 14};

    constexpr size_t NTMAX = (size_t)PB * PS * 32;   // 131072
    constexpr size_t BS    = (size_t)PB * PS;        // 4096

    // ---- carve workspace
    char* wsb = (char*)d_ws;
    size_t off = 0;
    auto carve = [&](size_t bytes) -> void* {
        void* p = wsb + off;
        off = (off + bytes + 255) & ~(size_t)255;
        return p;
    };
    int*      idxbuf = (int*)carve(NTMAX * sizeof(int));
    _Float16* w0a  = (_Float16*)carve(64  * 64  * 2);
    _Float16* w0b  = (_Float16*)carve(128 * 64  * 2);
    _Float16* w1a  = (_Float16*)carve(96  * 64  * 2);
    _Float16* w1b  = (_Float16*)carve(128 * 96  * 2);
    _Float16* wxyz = (_Float16*)carve(64  * 32  * 2);
    _Float16* wfc  = (_Float16*)carve(192 * 192 * 2);
    _Float16* wm   = (_Float16*)carve(128 * 192 * 2);
    _Float16* wfu  = (_Float16*)carve(128 * 128 * 2);
    float*    mean = (float*)carve(256 * sizeof(float));
    float*    var  = (float*)carve(256 * sizeof(float));
    _Float16* XTin   = (_Float16*)carve(NTMAX * 64  * 2);
    _Float16* XTxyz  = (_Float16*)carve(NTMAX * 32  * 2);
    _Float16* XTmid  = (_Float16*)carve(NTMAX * 96  * 2);
    _Float16* XTf    = (_Float16*)carve(NTMAX * 192 * 2);
    _Float16* aggXT  = (_Float16*)carve(BS * 192 * 2);
    _Float16* XTp    = (_Float16*)carve(BS * 128 * 2);
    float*    Y      = (float*)carve((size_t)192 * NTMAX * sizeof(float));
    if (off > ws_size) return;  // insufficient scratch

    float* out_xyz = (float*)d_out;                       // (B, S, 3)
    float* out_ftr = (float*)d_out + (size_t)PB * PS * 3; // (B, 256, S)

    auto bnact = [&](const float* Yi, const float* gg, const float* bb,
                     _Float16* XT, int M, int Ncols, int ld, int rowOff,
                     float ns, int doBN, int doAct) {
        size_t total = (size_t)M * Ncols;
        bnact_kernel<<<(unsigned)((total + 255) / 256), 256, 0, stream>>>(
            Yi, mean, var, gg, bb, XT, M, Ncols, ld, rowOff, ns, doBN, doAct);
    };

    // ---- weights to padded f16
    auto wcv = [&](int pi, _Float16* dst, int M, int K, int Kpad) {
        int tot = M * Kpad;
        wconv_kernel<<<(tot + 255) / 256, 256, 0, stream>>>((const float*)d_in[pi], dst, M, K, Kpad);
    };
    wcv(2, w0a, 64, 35, 64);
    wcv(6, w0b, 128, 64, 64);
    wcv(10, w1a, 96, 35, 64);
    wcv(14, w1b, 128, 96, 96);
    wcv(18, wxyz, 64, 10, 32);
    wcv(20, wfc, 192, 192, 192);
    wcv(21, wm, 128, 192, 192);
    wcv(25, wfu, 128, 128, 128);

    // ---- FPS -> new_xyz
    fps_kernel<<<PB, 1024, 0, stream>>>(xyz, out_xyz);

    const _Float16* WA[2] = {w0a, w1a};
    const _Float16* WB[2] = {w0b, w1b};

    for (int sc = 0; sc < 2; ++sc) {
        const int K = KS[sc];
        const int Cmid = CMID[sc];
        const int NT = PB * PS * K;
        const float r2 = RAD[sc] * RAD[sc];

        // ball query + grouping
        ballquery_kernel<<<(PB * PS + 255) / 256, 256, 0, stream>>>(xyz, out_xyz, idxbuf, r2, K);
        group_kernel<<<(NT + 255) / 256, 256, 0, stream>>>(xyz, feats, out_xyz, idxbuf,
                                                           XTin, XTxyz, K, NT);

        // MLP layer 1: (Cmid x 64) * XTin -> BN -> ReLU -> XTmid (ld=Cmid)
        launch_gemm(WA[sc], XTin, (const float*)d_in[L0W[sc] + 1], Y, Cmid, 64, NT, stream);
        stats_kernel<<<Cmid, 256, 0, stream>>>(Y, mean, var, NT);
        bnact(Y, (const float*)d_in[L0W[sc] + 2], (const float*)d_in[L0W[sc] + 3],
              XTmid, Cmid, NT, Cmid, 0, 0.f, 1, 1);

        // MLP layer 2: (128 x Cmid) * XTmid -> BN -> ReLU -> XTf rows 64..191 (ld=192)
        launch_gemm(WB[sc], XTmid, (const float*)d_in[L1W[sc] + 1], Y, 128, Cmid, NT, stream);
        stats_kernel<<<128, 256, 0, stream>>>(Y, mean, var, NT);
        bnact(Y, (const float*)d_in[L1W[sc] + 2], (const float*)d_in[L1W[sc] + 3],
              XTf, 128, NT, 192, 64, 0.f, 1, 1);

        // xyz conv: (64 x 32) * XTxyz + bxyz -> XTf rows 0..63 (no BN/act)
        launch_gemm(wxyz, XTxyz, (const float*)d_in[19], Y, 64, 32, NT, stream);
        bnact(Y, nullptr, nullptr, XTf, 64, NT, 192, 0, 0.f, 0, 0);

        // attention logits: (192 x 192) * XTf (no bias)
        launch_gemm(wfc, XTf, nullptr, Y, 192, 192, NT, stream);

        // softmax over K + weighted aggregation -> aggXT (BS x 192)
        attagg_kernel<<<(int)((BS * 192 + 255) / 256), 256, 0, stream>>>(Y, XTf, aggXT, NT, K);

        // Wm: (128 x 192) * aggXT + bm -> BN -> leaky_relu(0.01) -> XTp (BS x 128)
        launch_gemm(wm, aggXT, (const float*)d_in[22], Y, 128, 192, (int)BS, stream);
        stats_kernel<<<128, 256, 0, stream>>>(Y, mean, var, (int)BS);
        bnact(Y, (const float*)d_in[23], (const float*)d_in[24],
              XTp, 128, (int)BS, 128, 0, 0.01f, 1, 1);

        // fuse: (128 x 128) * XTp + b -> output channels [sc*128, sc*128+128)
        launch_gemm(wfu, XTp, (const float*)d_in[26], Y, 128, 128, (int)BS, stream);
        writeout_kernel<<<(int)((DOUT * BS + 255) / 256), 256, 0, stream>>>(Y, out_ftr, sc * 128);
    }
}
